// GraphTransformerPYG_34540126994706
// MI455X (gfx1250) — compile-verified, matrix operand order checked by reading, not run on hardware
//
#include <hip/hip_runtime.h>
#include <hip/hip_bf16.h>
#include <math.h>

// Problem constants (fixed by the reference).
#define NND 50000     // nodes (50000 = 3125 * 16, exact 16-row tiles)
#define NE  800000    // edges (= 50000 * 16)
#define CIN 128
#define HCC 128       // heads * hid
#define HIDD 32
#define NH  4
#define NGB 64        // graphs
#define NOUT 16

typedef __bf16 bf16;
typedef __attribute__((ext_vector_type(16))) __bf16 v16bf;
typedef __attribute__((ext_vector_type(8)))  __bf16 v8bf;
typedef __attribute__((ext_vector_type(8)))  float  v8f;

// ---------------------------------------------------------------------------
// fp32 -> (hi, lo) bf16 split. hi = rne(f), lo = rne(f - hi).
// ---------------------------------------------------------------------------
__global__ void k_split(const float* __restrict__ src, bf16* __restrict__ hi,
                        bf16* __restrict__ lo, int n) {
  int i = blockIdx.x * blockDim.x + threadIdx.x;
  if (i >= n) return;
  float f = src[i];
  bf16 h = (bf16)f;
  hi[i] = h;
  lo[i] = (bf16)(f - (float)h);
}

// Transposing split for weights: w is [K][C] row-major, out is [C][K].
__global__ void k_split_T(const float* __restrict__ w, bf16* __restrict__ thi,
                          bf16* __restrict__ tlo, int K, int C) {
  int i = blockIdx.x * blockDim.x + threadIdx.x;
  if (i >= K * C) return;
  int k = i / C, c = i - k * C;
  float f = w[i];
  bf16 h = (bf16)f;
  thi[c * K + k] = h;
  tlo[c * K + k] = (bf16)(f - (float)h);
}

__global__ void k_fill_f32(float* p, float v, int n) {
  int i = blockIdx.x * blockDim.x + threadIdx.x;
  if (i < n) p[i] = v;
}
__global__ void k_fill_u32(unsigned* p, unsigned v, int n) {
  int i = blockIdx.x * blockDim.x + threadIdx.x;
  if (i < n) p[i] = v;
}

// ---------------------------------------------------------------------------
// WMMA split-bf16 GEMM:  Y[M][128] = A[M][K] * B[K][128] + bias
// A supplied as hi/lo bf16 [M][K]; B supplied transposed hi/lo bf16 [128][K].
// One wave computes a 16-row strip across all 8 column tiles.
// Per 16x16x32 step: AhBh + AhBl + AlBh  (fp32-accurate split product).
// M must be a multiple of 16 (true here: 50000, 800000).
// ---------------------------------------------------------------------------
template <int K>
__global__ void k_wmma_gemm(const bf16* __restrict__ Ahi, const bf16* __restrict__ Alo,
                            const bf16* __restrict__ BThi, const bf16* __restrict__ BTlo,
                            const float* __restrict__ bias, float* __restrict__ Y, int M) {
  constexpr int KT = K / 32;
  int wv = blockIdx.x * (blockDim.x >> 5) + (threadIdx.x >> 5);
  if (wv >= (M >> 4)) return;          // wave-uniform: EXEC stays all-ones
  int lane = threadIdx.x & 31;
  int half = lane >> 4;
  int lrow = lane & 15;
  long r0 = (long)wv << 4;

  // A fragments: ISA 16-bit A layout. lane<16 -> K {0..7,16..23}, lane>=16 -> {8..15,24..31}.
  v16bf ahi[KT], alo[KT];
  const bf16* arh = Ahi + (r0 + lrow) * K;
  const bf16* arl = Alo + (r0 + lrow) * K;
#pragma unroll
  for (int kt = 0; kt < KT; ++kt) {
    int o0 = kt * 32 + half * 8;
    int o1 = o0 + 16;
    v8bf h0 = *(const v8bf*)(arh + o0);
    v8bf h1 = *(const v8bf*)(arh + o1);
    v8bf l0 = *(const v8bf*)(arl + o0);
    v8bf l1 = *(const v8bf*)(arl + o1);
#pragma unroll
    for (int j = 0; j < 8; ++j) {
      ahi[kt][j] = h0[j]; ahi[kt][8 + j] = h1[j];
      alo[kt][j] = l0[j]; alo[kt][8 + j] = l1[j];
    }
  }

#pragma unroll
  for (int ct = 0; ct < HCC / 16; ++ct) {
    int c0 = ct * 16;
    float bv = bias ? bias[c0 + lrow] : 0.0f;
    v8f acc;
#pragma unroll
    for (int j = 0; j < 8; ++j) acc[j] = bv;
    const bf16* brh = BThi + (long)(c0 + lrow) * K;
    const bf16* brl = BTlo + (long)(c0 + lrow) * K;
#pragma unroll
    for (int kt = 0; kt < KT; ++kt) {
      int o0 = kt * 32 + half * 8;
      int o1 = o0 + 16;
      v8bf h0 = *(const v8bf*)(brh + o0);
      v8bf h1 = *(const v8bf*)(brh + o1);
      v8bf l0 = *(const v8bf*)(brl + o0);
      v8bf l1 = *(const v8bf*)(brl + o1);
      v16bf bhi, blo;
#pragma unroll
      for (int j = 0; j < 8; ++j) {
        bhi[j] = h0[j]; bhi[8 + j] = h1[j];
        blo[j] = l0[j]; blo[8 + j] = l1[j];
      }
      acc = __builtin_amdgcn_wmma_f32_16x16x32_bf16(false, ahi[kt], false, bhi,
                                                    (short)0, acc, false, false);
      acc = __builtin_amdgcn_wmma_f32_16x16x32_bf16(false, ahi[kt], false, blo,
                                                    (short)0, acc, false, false);
      acc = __builtin_amdgcn_wmma_f32_16x16x32_bf16(false, alo[kt], false, bhi,
                                                    (short)0, acc, false, false);
    }
    // C/D layout: VGPR j -> row j (lanes 0-15) / row j+8 (lanes 16-31), col = lane&15.
#pragma unroll
    for (int j = 0; j < 8; ++j)
      Y[(r0 + half * 8 + j) * HCC + c0 + lrow] = acc[j];
  }
}

// ---------------------------------------------------------------------------
// Edge pass A: alpha[e,h] = q[tgt,h,:]·(k[src,h,:]+e[e,h,:]) / sqrt(32);
// segment-max into ordered-uint encoded maxEnc[tgt,h].
// ---------------------------------------------------------------------------
__device__ __forceinline__ unsigned ord_enc(float f) {
  unsigned u = __float_as_uint(f);
  return (u & 0x80000000u) ? ~u : (u | 0x80000000u);
}
__device__ __forceinline__ float ord_dec(unsigned u) {
  return (u & 0x80000000u) ? __uint_as_float(u ^ 0x80000000u) : __uint_as_float(~u);
}

__global__ void k_edge_alpha(const int* __restrict__ src, const int* __restrict__ tgt,
                             const float* __restrict__ qn, const float* __restrict__ kn,
                             const float* __restrict__ ebuf, float* __restrict__ alpha,
                             unsigned* __restrict__ maxEnc) {
  int idx = blockIdx.x * blockDim.x + threadIdx.x;
  if (idx >= NE * NH) return;
  int e = idx >> 2, h = idx & 3;
  int s = src[e], t = tgt[e];
  const float4* qp = (const float4*)(qn + (long)t * HCC + h * HIDD);
  const float4* kp = (const float4*)(kn + (long)s * HCC + h * HIDD);
  const float4* ep = (const float4*)(ebuf + (long)e * HCC + h * HIDD);
  float acc = 0.0f;
#pragma unroll
  for (int i = 0; i < HIDD / 4; ++i) {
    float4 q4 = qp[i], k4 = kp[i], e4 = ep[i];
    acc += q4.x * (k4.x + e4.x) + q4.y * (k4.y + e4.y) +
           q4.z * (k4.z + e4.z) + q4.w * (k4.w + e4.w);
  }
  float a = acc * 0.17677669529663687f; // 1/sqrt(32)
  alpha[idx] = a;
  atomicMax(maxEnc + (long)t * NH + h, ord_enc(a));
}

// Edge pass B: a = exp(alpha - max[tgt]); den[tgt] += a (alpha overwritten with a).
__global__ void k_edge_exp(const int* __restrict__ tgt, float* __restrict__ alpha,
                           const unsigned* __restrict__ maxEnc, float* __restrict__ den) {
  int idx = blockIdx.x * blockDim.x + threadIdx.x;
  if (idx >= NE * NH) return;
  int e = idx >> 2, h = idx & 3;
  int t = tgt[e];
  float m = ord_dec(maxEnc[(long)t * NH + h]);
  float a = __expf(alpha[idx] - m);
  alpha[idx] = a;
  atomicAdd(den + (long)t * NH + h, a);
}

// Edge pass C: agg[tgt] += (a/den[tgt]) * (v[src]+e).  agg pre-initialized with skip.
__global__ void k_edge_agg(const int* __restrict__ src, const int* __restrict__ tgt,
                           const float* __restrict__ vn, const float* __restrict__ ebuf,
                           const float* __restrict__ alpha, const float* __restrict__ den,
                           float* __restrict__ agg) {
  int idx = blockIdx.x * blockDim.x + threadIdx.x;
  if (idx >= NE * NH) return;
  int e = idx >> 2, h = idx & 3;
  int s = src[e], t = tgt[e];
  float w = alpha[idx] / den[(long)t * NH + h];
  const float4* vp = (const float4*)(vn + (long)s * HCC + h * HIDD);
  const float4* ep = (const float4*)(ebuf + (long)e * HCC + h * HIDD);
  float* ag = agg + (long)t * HCC + h * HIDD;
#pragma unroll
  for (int i = 0; i < HIDD / 4; ++i) {
    float4 v4 = vp[i], e4 = ep[i];
    atomicAdd(ag + i * 4 + 0, w * (v4.x + e4.x));
    atomicAdd(ag + i * 4 + 1, w * (v4.y + e4.y));
    atomicAdd(ag + i * 4 + 2, w * (v4.z + e4.z));
    atomicAdd(ag + i * 4 + 3, w * (v4.w + e4.w));
  }
}

__global__ void k_elu(float* p, int n) {
  int i = blockIdx.x * blockDim.x + threadIdx.x;
  if (i >= n) return;
  float f = p[i];
  p[i] = f > 0.0f ? f : (__expf(f) - 1.0f);
}

// Mean-pool: pooled[b] += h[n]; one thread per (node, 4-col group).
__global__ void k_pool(const float* __restrict__ h, const int* __restrict__ batch,
                       float* __restrict__ pooled) {
  int idx = blockIdx.x * blockDim.x + threadIdx.x;
  if (idx >= NND * (HCC / 4)) return;
  int n = idx >> 5, g = idx & 31;
  int b = batch[n];
  float4 v = *(const float4*)(h + (long)n * HCC + g * 4);
  float* pp = pooled + (long)b * HCC + g * 4;
  atomicAdd(pp + 0, v.x);
  atomicAdd(pp + 1, v.y);
  atomicAdd(pp + 2, v.z);
  atomicAdd(pp + 3, v.w);
}

__global__ void k_count(const int* __restrict__ batch, float* __restrict__ cnt) {
  int i = blockIdx.x * blockDim.x + threadIdx.x;
  if (i < NND) atomicAdd(cnt + batch[i], 1.0f);
}

// Classifier + log_softmax. One block (32 lanes) per graph; lanes 0..15 own a class.
__global__ void k_head(const float* __restrict__ pooled, const float* __restrict__ cnt,
                       const float* __restrict__ wlin, const float* __restrict__ blin,
                       float* __restrict__ out) {
  int b = blockIdx.x;
  int l = threadIdx.x;
  float cf = fmaxf(cnt[b], 1.0f);
  float logit = -INFINITY;
  if (l < NOUT) {
    float acc = blin[l];
    for (int c = 0; c < HCC; ++c)
      acc += (pooled[(long)b * HCC + c] / cf) * wlin[c * NOUT + l];
    logit = acc;
  }
  float m = logit;
  for (int o = 8; o >= 1; o >>= 1) m = fmaxf(m, __shfl_xor(m, o, 16));
  float ex = (l < NOUT) ? __expf(logit - m) : 0.0f;
  float s = ex;
  for (int o = 8; o >= 1; o >>= 1) s += __shfl_xor(s, o, 16);
  if (l < NOUT) out[(long)b * NOUT + l] = logit - m - __logf(s);
}

// ---------------------------------------------------------------------------
// Host orchestration
// ---------------------------------------------------------------------------
static inline int gb(long n, int t) { return (int)((n + t - 1) / t); }

extern "C" void kernel_launch(void* const* d_in, const int* in_sizes, int n_in,
                              void* d_out, int out_size, void* d_ws, size_t ws_size,
                              hipStream_t stream) {
  (void)in_sizes; (void)n_in; (void)out_size; (void)ws_size;
  const float* x    = (const float*)d_in[0];
  const int*   ei   = (const int*)d_in[1];
  const float* ea   = (const float*)d_in[2];
  const int*   bat  = (const int*)d_in[3];
  const float* W[2][5] = {{(const float*)d_in[4],  (const float*)d_in[6],
                           (const float*)d_in[8],  (const float*)d_in[10],
                           (const float*)d_in[11]},
                          {(const float*)d_in[13], (const float*)d_in[15],
                           (const float*)d_in[17], (const float*)d_in[19],
                           (const float*)d_in[20]}};
  // W[l] = {wq, wk, wv, we, ws}
  const float* Bv[2][4] = {{(const float*)d_in[5],  (const float*)d_in[7],
                            (const float*)d_in[9],  (const float*)d_in[12]},
                           {(const float*)d_in[14], (const float*)d_in[16],
                            (const float*)d_in[18], (const float*)d_in[21]}};
  // Bv[l] = {bq, bk, bv, bs}
  const float* wlin = (const float*)d_in[22];
  const float* blin = (const float*)d_in[23];
  float* out = (float*)d_out;

  const int* src = ei;
  const int* tgt = ei + NE;

  // --- workspace bump allocator (all sizes 256B-aligned) ---
  char* p = (char*)d_ws;
  auto alloc = [&](size_t bytes) { char* r = p; p += (bytes + 255) & ~(size_t)255; return r; };
  bf16* xhi  = (bf16*)alloc((size_t)NND * HCC * 2);
  bf16* xlo  = (bf16*)alloc((size_t)NND * HCC * 2);
  bf16* eahi = (bf16*)alloc((size_t)NE * HIDD * 2);
  bf16* ealo = (bf16*)alloc((size_t)NE * HIDD * 2);
  float* qn  = (float*)alloc((size_t)NND * HCC * 4);
  float* kn  = (float*)alloc((size_t)NND * HCC * 4);
  float* vn  = (float*)alloc((size_t)NND * HCC * 4);
  float* aggA = (float*)alloc((size_t)NND * HCC * 4);
  float* aggB = (float*)alloc((size_t)NND * HCC * 4);
  float* ebuf = (float*)alloc((size_t)NE * HCC * 4);
  float* alpha = (float*)alloc((size_t)NE * NH * 4);
  unsigned* maxEnc = (unsigned*)alloc((size_t)NND * NH * 4);
  float* den = (float*)alloc((size_t)NND * NH * 4);
  float* pooled = (float*)alloc((size_t)NGB * HCC * 4);
  float* cnt = (float*)alloc((size_t)NGB * 4);
  // Converted (transposed, hi/lo split) weights: [layer][mat 0..4] with K = 128 except we(K=32).
  bf16* wThi[2][5];
  bf16* wTlo[2][5];
  for (int l = 0; l < 2; ++l)
    for (int m = 0; m < 5; ++m) {
      int K = (m == 3) ? HIDD : HCC;
      wThi[l][m] = (bf16*)alloc((size_t)K * HCC * 2);
      wTlo[l][m] = (bf16*)alloc((size_t)K * HCC * 2);
    }

  const int T = 256;

  // --- one-time conversions (re-done every call: deterministic) ---
  for (int l = 0; l < 2; ++l)
    for (int m = 0; m < 5; ++m) {
      int K = (m == 3) ? HIDD : HCC;
      k_split_T<<<gb((long)K * HCC, T), T, 0, stream>>>(W[l][m], wThi[l][m], wTlo[l][m], K, HCC);
    }
  k_split<<<gb((long)NE * HIDD, T), T, 0, stream>>>(ea, eahi, ealo, NE * HIDD);

  const int nodeStrips = NND / 16, edgeStrips = NE / 16;
  const int nodeBlocks = gb(nodeStrips, 8), edgeBlocks = gb(edgeStrips, 8);
  const long EH = (long)NE * NH;

  // --- per-layer routine ---
  auto layer = [&](const float* xin, int l, float* agg) {
    // split input features
    k_split<<<gb((long)NND * HCC, T), T, 0, stream>>>(xin, xhi, xlo, NND * HCC);
    // node GEMMs (q, k, v, skip->agg) — WMMA
    k_wmma_gemm<HCC><<<nodeBlocks, T, 0, stream>>>(xhi, xlo, wThi[l][0], wTlo[l][0], Bv[l][0], qn, NND);
    k_wmma_gemm<HCC><<<nodeBlocks, T, 0, stream>>>(xhi, xlo, wThi[l][1], wTlo[l][1], Bv[l][1], kn, NND);
    k_wmma_gemm<HCC><<<nodeBlocks, T, 0, stream>>>(xhi, xlo, wThi[l][2], wTlo[l][2], Bv[l][2], vn, NND);
    k_wmma_gemm<HCC><<<nodeBlocks, T, 0, stream>>>(xhi, xlo, wThi[l][4], wTlo[l][4], Bv[l][3], agg, NND);
    // edge embedding GEMM (E x 32 @ 32 x 128) — WMMA
    k_wmma_gemm<HIDD><<<edgeBlocks, T, 0, stream>>>(eahi, ealo, wThi[l][3], wTlo[l][3], nullptr, ebuf, NE);
    // segment softmax + aggregation
    k_fill_u32<<<gb((long)NND * NH, T), T, 0, stream>>>(maxEnc, 0x007FFFFFu /* enc(-inf) */, NND * NH);
    k_fill_f32<<<gb((long)NND * NH, T), T, 0, stream>>>(den, 0.0f, NND * NH);
    k_edge_alpha<<<gb(EH, T), T, 0, stream>>>(src, tgt, qn, kn, ebuf, alpha, maxEnc);
    k_edge_exp<<<gb(EH, T), T, 0, stream>>>(tgt, alpha, maxEnc, den);
    k_edge_agg<<<gb(EH, T), T, 0, stream>>>(src, tgt, vn, ebuf, alpha, den, agg);
    k_elu<<<gb((long)NND * HCC, T), T, 0, stream>>>(agg, NND * HCC);
  };

  layer(x, 0, aggA);     // h1 = elu(conv1(x))   in aggA
  layer(aggA, 1, aggB);  // h2 = elu(conv2(h1))  in aggB

  // --- global mean pool + classifier + log_softmax ---
  k_fill_f32<<<gb((long)NGB * HCC, T), T, 0, stream>>>(pooled, 0.0f, NGB * HCC);
  k_fill_f32<<<gb((long)NGB, T), T, 0, stream>>>(cnt, 0.0f, NGB);
  k_pool<<<gb((long)NND * (HCC / 4), T), T, 0, stream>>>(aggB, bat, pooled);
  k_count<<<gb((long)NND, T), T, 0, stream>>>(bat, cnt);
  k_head<<<NGB, 32, 0, stream>>>(pooled, cnt, wlin, blin, out);
}
